// tblock_86629490360601
// MI455X (gfx1250) — matrix operand order repaired and checked by reading, not compiled, not run on hardware
//
#include <hip/hip_runtime.h>
#include <hip/hip_bf16.h>
#include <math.h>

// ---------------- model constants ----------------
#define BATCH    2
#define SEQ      2048
#define DMODEL   1024
#define NHEADS   16
#define DHEAD    64
#define NGROUPS  8
#define DMLP     4096
#define ROWS     (BATCH * SEQ)        // 4096
#define QKV_LD   2048                 // Q(1024) | K(512) | V(512)

typedef __bf16 bf16_t;
typedef __attribute__((ext_vector_type(16))) __bf16 v16bf;
typedef __attribute__((ext_vector_type(8)))  __bf16 v8bf;
typedef __attribute__((ext_vector_type(8)))  float  v8f;

union V16U { v16bf v; v8bf h[2]; __bf16 e[16]; };

// =====================================================================
// Weight repack: float W -> bf16 Bop[kt][n][ki]  (kt = k/32, ki = k%32)
// mode 0: W stored [K][N] row-major (W_Q / W_K / W_V)
// mode 1: W stored [N][K] row-major (mlp1_w / mlp2_w, used as x @ W^T)
// mode 2: W_out stored [dk(64)][h(16)][d(1024)], reduction index = h*64+dk
// =====================================================================
__global__ __launch_bounds__(256)
void repack_kernel(const float* __restrict__ W, bf16_t* __restrict__ Bop,
                   int K, int N, int mode)
{
    size_t idx = (size_t)blockIdx.x * blockDim.x + threadIdx.x;
    size_t total = (size_t)K * N;
    if (idx >= total) return;
    int ki = (int)(idx & 31);
    int t  = (int)(idx >> 5);
    int n  = t % N;
    int kt = t / N;
    int k  = kt * 32 + ki;
    float v;
    if (mode == 0)      v = W[(size_t)k * N + n];
    else if (mode == 1) v = W[(size_t)n * K + k];
    else { int hh = k >> 6, dk = k & 63; v = W[((size_t)dk * NHEADS + hh) * DMODEL + n]; }
    Bop[idx] = (bf16_t)v;
}

// =====================================================================
// LayerNorm (f32 in -> bf16 out), one row (1024) per block
// =====================================================================
__global__ __launch_bounds__(256)
void ln_kernel(const float* __restrict__ x, const float* __restrict__ w,
               const float* __restrict__ b, bf16_t* __restrict__ out)
{
    __shared__ float red[256];
    __shared__ float red2[256];
    const int row = blockIdx.x;
    const float* xr = x + (size_t)row * DMODEL;
    float s = 0.f, s2 = 0.f;
    for (int c = threadIdx.x; c < DMODEL; c += 256) {
        float v = xr[c]; s += v; s2 += v * v;
    }
    red[threadIdx.x] = s; red2[threadIdx.x] = s2;
    __syncthreads();
    for (int off = 128; off > 0; off >>= 1) {
        if (threadIdx.x < off) {
            red[threadIdx.x]  += red[threadIdx.x + off];
            red2[threadIdx.x] += red2[threadIdx.x + off];
        }
        __syncthreads();
    }
    float mu  = red[0]  * (1.0f / DMODEL);
    float var = red2[0] * (1.0f / DMODEL) - mu * mu;
    float rs  = rsqrtf(var + 1e-5f);
    for (int c = threadIdx.x; c < DMODEL; c += 256)
        out[(size_t)row * DMODEL + c] = (bf16_t)((xr[c] - mu) * rs * w[c] + b[c]);
}

// =====================================================================
// V transpose: qkv[row][1024 + g*64 + dk] -> Vt[(b,g,dk)][t]
// =====================================================================
__global__ __launch_bounds__(256)
void vt_kernel(const bf16_t* __restrict__ qkv, bf16_t* __restrict__ Vt)
{
    size_t idx = (size_t)blockIdx.x * blockDim.x + threadIdx.x;
    if (idx >= (size_t)BATCH * NGROUPS * DHEAD * SEQ) return;
    int t  = (int)(idx & (SEQ - 1));
    size_t r = idx >> 11;
    int dk = (int)(r & 63); r >>= 6;
    int g  = (int)(r & 7);
    int b  = (int)(r >> 3);
    Vt[idx] = qkv[((size_t)(b * SEQ + t)) * QKV_LD + 1024 + g * 64 + dk];
}

// =====================================================================
// Generic bf16 WMMA GEMM: C[M,N] = A[M,K](bf16,row-major) x Bop (repacked)
// block = 256 threads = 8 waves (2x4); wave tile 32x64; block tile 64x256
// epilogue: +bias, GELU(erf), +residual(f32), store f32 and/or bf16
// =====================================================================
__global__ __launch_bounds__(256)
void gemm_kernel(const bf16_t* __restrict__ A, const bf16_t* __restrict__ Bop,
                 float* __restrict__ outf, bf16_t* __restrict__ outb, int ldo,
                 const float* __restrict__ bias, const float* __restrict__ resid,
                 int M, int N, int K, int act)
{
    const int lane = threadIdx.x & 31;
    const int wave = threadIdx.x >> 5;
    const int wm = wave >> 2, wn = wave & 3;
    const int half = lane >> 4, l15 = lane & 15;
    const int m0 = blockIdx.y * 64  + wm * 32;
    const int n0 = blockIdx.x * 256 + wn * 64;
    const int kb = half * 16;

    v8f acc[2][4] = {};

    const int ktiles = K >> 5;
#pragma unroll 2
    for (int kt = 0; kt < ktiles; ++kt) {
        // Straight-line WGP-scope prefetch of the next K-step (branchless
        // clamp keeps the address inside the mapped workspace; lowers to
        // global_prefetch_b8).
        const int ktn = (kt + 1 < ktiles) ? (kt + 1) : kt;
        __builtin_prefetch(A + (size_t)(m0 + l15) * K + ktn * 32, 0, 3);
        __builtin_prefetch(Bop + ((size_t)ktn * N + (n0 + lane * 2)) * 32, 0, 3);

        V16U a[2];
#pragma unroll
        for (int i = 0; i < 2; ++i) {
            const bf16_t* ap = A + (size_t)(m0 + i * 16 + l15) * K + kt * 32 + kb;
            a[i].h[0] = *(const v8bf*)(ap);
            a[i].h[1] = *(const v8bf*)(ap + 8);
        }
#pragma unroll
        for (int j = 0; j < 4; ++j) {
            const bf16_t* bp = Bop + ((size_t)kt * N + (n0 + j * 16 + l15)) * 32 + half * 16;
            v16bf bv = *(const v16bf*)bp;
#pragma unroll
            for (int i = 0; i < 2; ++i)
                acc[i][j] = __builtin_amdgcn_wmma_f32_16x16x32_bf16(
                    false, a[i].v, false, bv, (short)0, acc[i][j], false, false);
        }
    }

#pragma unroll
    for (int i = 0; i < 2; ++i) {
#pragma unroll
        for (int j = 0; j < 4; ++j) {
#pragma unroll
            for (int r = 0; r < 8; ++r) {
                int row = m0 + i * 16 + r + half * 8;
                int col = n0 + j * 16 + l15;
                float v = acc[i][j][r];
                if (bias)  v += bias[col];
                if (act)   v = 0.5f * v * (1.0f + erff(v * 0.70710678118654752f));
                if (resid) v += resid[(size_t)row * ldo + col];
                if (outf)  outf[(size_t)row * ldo + col] = v;
                if (outb)  outb[(size_t)row * ldo + col] = (bf16_t)v;
            }
        }
    }
}

// =====================================================================
// Flash-style GQA causal attention. One wave = 16 queries of one head.
// Processes 32 keys per iteration: 4 score WMMAs + 4 full-depth (K=32)
// PV WMMAs -- no zero-padding waste. Online softmax in registers.
// grid = (SEQ/16, NHEADS, BATCH), block = 32 threads.
// =====================================================================
__global__ __launch_bounds__(32)
void attn_kernel(const bf16_t* __restrict__ qkv, const bf16_t* __restrict__ Vt,
                 bf16_t* __restrict__ z)
{
    __shared__ bf16_t ldsP[16 * 32];      // P tile: 16 rows x 32 keys
    const int lane = threadIdx.x;
    const int qt = blockIdx.x, h = blockIdx.y, b = blockIdx.z;
    const int g = h >> 1;                  // GROUP_SIZE = 2
    const int q0 = qt * 16;
    const int half = lane >> 4;
    const int l15  = lane & 15;
    const int kb   = half * 16;

    // Q tile as two A operands (head-dim chunks 0..31 / 32..63)
    V16U aq[2];
    {
        const bf16_t* qp = qkv + ((size_t)(b * SEQ + q0 + l15)) * QKV_LD + h * 64;
#pragma unroll
        for (int kc = 0; kc < 2; ++kc) {
            aq[kc].h[0] = *(const v8bf*)(qp + kc * 32 + kb);
            aq[kc].h[1] = *(const v8bf*)(qp + kc * 32 + kb + 8);
        }
    }

    v8f o[4] = {};
    float mrow[8], lrow[8];
#pragma unroll
    for (int r = 0; r < 8; ++r) { mrow[r] = -3.0e38f; lrow[r] = 0.f; }

    const bf16_t* vbase = Vt + ((size_t)((b * NGROUPS + g) * DHEAD)) * SEQ;

    // pairs of 16-key tiles; overshoot tile (qt+1, when qt even) is fully
    // masked and always in range since SEQ/16 = 128 is even.
    const int npairs = (qt + 2) >> 1;
    for (int ip = 0; ip < npairs; ++ip) {
        const int t0tile = 2 * ip, t1tile = 2 * ip + 1;

        // K^T as B operands: per-lane contiguous 16 halves of a K-row
        V16U bk[2][2];
#pragma unroll
        for (int ti = 0; ti < 2; ++ti) {
            const bf16_t* kp = qkv
                + ((size_t)(b * SEQ + (t0tile + ti) * 16 + l15)) * QKV_LD
                + 1024 + g * 64 + half * 16;
            bk[ti][0].v = *(const v16bf*)(kp);
            bk[ti][1].v = *(const v16bf*)(kp + 32);
        }
        v8f s0 = {}, s1 = {};
        s0 = __builtin_amdgcn_wmma_f32_16x16x32_bf16(false, aq[0].v, false, bk[0][0].v, (short)0, s0, false, false);
        s0 = __builtin_amdgcn_wmma_f32_16x16x32_bf16(false, aq[1].v, false, bk[0][1].v, (short)0, s0, false, false);
        s1 = __builtin_amdgcn_wmma_f32_16x16x32_bf16(false, aq[0].v, false, bk[1][0].v, (short)0, s1, false, false);
        s1 = __builtin_amdgcn_wmma_f32_16x16x32_bf16(false, aq[1].v, false, bk[1][1].v, (short)0, s1, false, false);

        const bool chk0 = (t0tile >= qt), chk1 = (t1tile >= qt);
        float p0[8], p1[8], corr[8];
#pragma unroll
        for (int r = 0; r < 8; ++r) {
            float v0 = s0[r] * 0.125f;             // 1/sqrt(64)
            float v1 = s1[r] * 0.125f;
            const int qrow = q0 + r + half * 8;
            if (chk0 && (t0tile * 16 + l15 > qrow)) v0 = -1.0e6f;
            if (chk1 && (t1tile * 16 + l15 > qrow)) v1 = -1.0e6f;
            float mx = fmaxf(v0, v1);
            mx = fmaxf(mx, __shfl_xor(mx, 1, 16));
            mx = fmaxf(mx, __shfl_xor(mx, 2, 16));
            mx = fmaxf(mx, __shfl_xor(mx, 4, 16));
            mx = fmaxf(mx, __shfl_xor(mx, 8, 16));
            float mn = fmaxf(mrow[r], mx);
            float c  = __expf(mrow[r] - mn);
            float pe0 = __expf(v0 - mn);
            float pe1 = __expf(v1 - mn);
            float rs = pe0 + pe1;
            rs += __shfl_xor(rs, 1, 16);
            rs += __shfl_xor(rs, 2, 16);
            rs += __shfl_xor(rs, 4, 16);
            rs += __shfl_xor(rs, 8, 16);
            lrow[r] = lrow[r] * c + rs;
            mrow[r] = mn;
            corr[r] = c;
            p0[r] = pe0;
            p1[r] = pe1;
        }
#pragma unroll
        for (int r = 0; r < 8; ++r) {
            o[0][r] *= corr[r]; o[1][r] *= corr[r];
            o[2][r] *= corr[r]; o[3][r] *= corr[r];
        }

        // P (two 16x16 tiles): C/D layout -> 16x32 A-operand layout via LDS
        __syncthreads();
#pragma unroll
        for (int r = 0; r < 8; ++r) {
            ldsP[(r + half * 8) * 32 + l15]      = (bf16_t)p0[r];
            ldsP[(r + half * 8) * 32 + 16 + l15] = (bf16_t)p1[r];
        }
        __syncthreads();

        V16U ap;
#pragma unroll
        for (int e = 0; e < 16; ++e) {
            const int kk = kb + e;   // A 16x32 element map
            ap.e[e] = ldsP[l15 * 32 + kk];
        }

        // V as full-depth (K=32) B operand: contiguous 16 halves per lane
        const bf16_t* vp = vbase + ip * 32 + half * 16;
#pragma unroll
        for (int j = 0; j < 4; ++j) {
            v16bf bv = *(const v16bf*)(vp + (size_t)(j * 16 + l15) * SEQ);
            o[j] = __builtin_amdgcn_wmma_f32_16x16x32_bf16(
                false, ap.v, false, bv, (short)0, o[j], false, false);
        }
        __syncthreads();
    }

#pragma unroll
    for (int r = 0; r < 8; ++r) {
        float inv = 1.0f / lrow[r];
        int row = q0 + r + half * 8;
        bf16_t* zp = z + ((size_t)(b * SEQ + row)) * DMODEL + h * 64 + l15;
        zp[0]  = (bf16_t)(o[0][r] * inv);
        zp[16] = (bf16_t)(o[1][r] * inv);
        zp[32] = (bf16_t)(o[2][r] * inv);
        zp[48] = (bf16_t)(o[3][r] * inv);
    }
}

// =====================================================================
// Host-side orchestration
// =====================================================================
extern "C" void kernel_launch(void* const* d_in, const int* in_sizes, int n_in,
                              void* d_out, int out_size, void* d_ws, size_t ws_size,
                              hipStream_t stream)
{
    const float* x      = (const float*)d_in[0];
    const float* ln1_w  = (const float*)d_in[1];
    const float* ln1_b  = (const float*)d_in[2];
    const float* W_Q    = (const float*)d_in[3];
    const float* W_K    = (const float*)d_in[4];
    const float* W_V    = (const float*)d_in[5];
    const float* W_out  = (const float*)d_in[6];
    const float* b_out  = (const float*)d_in[7];
    const float* ln2_w  = (const float*)d_in[8];
    const float* ln2_b  = (const float*)d_in[9];
    const float* mlp1_w = (const float*)d_in[10];
    const float* mlp1_b = (const float*)d_in[11];
    const float* mlp2_w = (const float*)d_in[12];
    const float* mlp2_b = (const float*)d_in[13];
    float* out = (float*)d_out;

    // workspace carve-up (256B aligned)
    size_t off = 0;
    char* base = (char*)d_ws;
    auto carve = [&](size_t bytes) -> char* {
        char* p = base + off;
        off += (bytes + 255) & ~(size_t)255;
        return p;
    };
    bf16_t* normed1  = (bf16_t*)carve((size_t)ROWS * DMODEL * 2);      // 8 MB
    bf16_t* qkv      = (bf16_t*)carve((size_t)ROWS * QKV_LD * 2);      // 16 MB
    bf16_t* Vt       = (bf16_t*)carve((size_t)BATCH * NGROUPS * DHEAD * SEQ * 2); // 4 MB
    bf16_t* zmat     = (bf16_t*)carve((size_t)ROWS * DMODEL * 2);      // 8 MB
    float*  post     = (float*) carve((size_t)ROWS * DMODEL * 4);      // 16 MB
    bf16_t* normed2  = (bf16_t*)carve((size_t)ROWS * DMODEL * 2);      // 8 MB
    bf16_t* hact     = (bf16_t*)carve((size_t)ROWS * DMLP   * 2);      // 32 MB
    bf16_t* Bq       = (bf16_t*)carve((size_t)DMODEL * DMODEL * 2);    // 2 MB
    bf16_t* Bk       = (bf16_t*)carve((size_t)DMODEL * (NGROUPS*DHEAD) * 2);
    bf16_t* Bv       = (bf16_t*)carve((size_t)DMODEL * (NGROUPS*DHEAD) * 2);
    bf16_t* Bo       = (bf16_t*)carve((size_t)DMODEL * DMODEL * 2);
    bf16_t* B1       = (bf16_t*)carve((size_t)DMODEL * DMLP * 2);      // 8 MB
    bf16_t* B2       = (bf16_t*)carve((size_t)DMLP * DMODEL * 2);      // 8 MB
    (void)ws_size; (void)n_in; (void)in_sizes; (void)out_size;

    auto blocks = [](size_t total) { return (unsigned)((total + 255) / 256); };

    // 1) repack weights to bf16 WMMA-B layout
    repack_kernel<<<blocks((size_t)DMODEL*DMODEL), 256, 0, stream>>>(W_Q,    Bq, DMODEL, DMODEL, 0);
    repack_kernel<<<blocks((size_t)DMODEL*512),    256, 0, stream>>>(W_K,    Bk, DMODEL, 512,    0);
    repack_kernel<<<blocks((size_t)DMODEL*512),    256, 0, stream>>>(W_V,    Bv, DMODEL, 512,    0);
    repack_kernel<<<blocks((size_t)DMODEL*DMODEL), 256, 0, stream>>>(W_out,  Bo, DMODEL, DMODEL, 2);
    repack_kernel<<<blocks((size_t)DMODEL*DMLP),   256, 0, stream>>>(mlp1_w, B1, DMODEL, DMLP,   1);
    repack_kernel<<<blocks((size_t)DMLP*DMODEL),   256, 0, stream>>>(mlp2_w, B2, DMLP,   DMODEL, 1);

    // 2) LN1
    ln_kernel<<<ROWS, 256, 0, stream>>>(x, ln1_w, ln1_b, normed1);

    // 3) QKV projections (bf16 outputs packed into qkv rows of 2048)
    gemm_kernel<<<dim3(DMODEL/256, ROWS/64), 256, 0, stream>>>(
        normed1, Bq, nullptr, qkv,        QKV_LD, nullptr, nullptr, ROWS, DMODEL, DMODEL, 0);
    gemm_kernel<<<dim3(512/256,  ROWS/64), 256, 0, stream>>>(
        normed1, Bk, nullptr, qkv + 1024, QKV_LD, nullptr, nullptr, ROWS, 512, DMODEL, 0);
    gemm_kernel<<<dim3(512/256,  ROWS/64), 256, 0, stream>>>(
        normed1, Bv, nullptr, qkv + 1536, QKV_LD, nullptr, nullptr, ROWS, 512, DMODEL, 0);

    // 4) transpose V for PV B-operand
    vt_kernel<<<blocks((size_t)BATCH*NGROUPS*DHEAD*SEQ), 256, 0, stream>>>(qkv, Vt);

    // 5) flash attention (causal, GQA)
    attn_kernel<<<dim3(SEQ/16, NHEADS, BATCH), 32, 0, stream>>>(qkv, Vt, zmat);

    // 6) output projection + b_out + residual x  -> post_attn (f32)
    gemm_kernel<<<dim3(DMODEL/256, ROWS/64), 256, 0, stream>>>(
        zmat, Bo, post, nullptr, DMODEL, b_out, x, ROWS, DMODEL, DMODEL, 0);

    // 7) LN2
    ln_kernel<<<ROWS, 256, 0, stream>>>(post, ln2_w, ln2_b, normed2);

    // 8) MLP1 + bias + GELU(erf) -> bf16
    gemm_kernel<<<dim3(DMLP/256, ROWS/64), 256, 0, stream>>>(
        normed2, B1, nullptr, hact, DMLP, mlp1_b, nullptr, ROWS, DMLP, DMODEL, 1);

    // 9) MLP2 + bias + residual post_attn -> d_out (f32)
    gemm_kernel<<<dim3(DMODEL/256, ROWS/64), 256, 0, stream>>>(
        hact, B2, out, nullptr, DMODEL, mlp2_b, post, ROWS, DMODEL, DMLP, 0);
}
